// general_RNN_83854941487518
// MI455X (gfx1250) — matrix operand order repaired and checked by reading, not compiled, not run on hardware
//
#include <hip/hip_runtime.h>
#include <stdint.h>

// Problem constants (match reference)
#define V_SIZE    512
#define FULL_SIZE 1024
#define T_STEPS   128

// GEMM tiling
#define BM 64
#define BN 64
#define BK 64
#define PAD 8                       // 16B row padding -> bank spread (TDM pad matches)
#define KTILES (FULL_SIZE / BK)     // 16 double-buffered K iterations

typedef __bf16 bf16_t;
typedef __bf16 v16bf __attribute__((ext_vector_type(16)));
typedef __bf16 v8bf  __attribute__((ext_vector_type(8)));
typedef float  v8f   __attribute__((ext_vector_type(8)));
typedef unsigned int u32x4 __attribute__((ext_vector_type(4)));
typedef int          i32x4 __attribute__((ext_vector_type(4)));
typedef int          i32x8 __attribute__((ext_vector_type(8)));

#if __has_builtin(__builtin_amdgcn_tensor_load_to_lds) && \
    __has_builtin(__builtin_amdgcn_s_wait_tensorcnt)
#define USE_TDM 1
#else
#define USE_TDM 0
#endif

// ---- CDNA5 async global->LDS copy (ASYNCcnt-tracked, bypasses VGPRs) ----
__device__ __forceinline__ void async_b128(uint32_t lds_off, const void* gptr) {
    asm volatile("global_load_async_to_lds_b128 %0, %1, off"
                 :: "v"(lds_off), "v"(gptr) : "memory");
}
#define WAIT_ASYNCCNT(n) asm volatile("s_wait_asynccnt " #n ::: "memory")

#if USE_TDM
// ---- TDM: one descriptor moves a 64x64 bf16 tile (2048B row pitch) into LDS,
//      padding 4 DWORDs after every 32 DWORDs => 144B LDS row stride. ----
__device__ __forceinline__ void tdm_load_tile(const bf16_t* gtile, uint32_t lds_addr,
                                              int tensor_rows) {
    const uint64_t ga = (uint64_t)(uintptr_t)gtile;
    const u32x4 g0 = {
        1u,                                                   // count=1 (valid D#)
        lds_addr,                                             // lds_addr [63:32]
        (uint32_t)ga,                                         // global_addr lo
        (uint32_t)((ga >> 32) & 0x01ffffffu) | (2u << 30)     // addr hi | type=2
    };
    const i32x8 g1 = {
        (int)((1u << 16) | (1u << 20) | (4u << 22) | (3u << 25)),
        // data_size=2B | pad_enable | pad_interval=32dw | pad_amount=4dw
        (int)(1024u << 16),               // tensor_dim0 = 1024 (bits 63:48)
        (int)((uint32_t)tensor_rows << 16), // tensor_dim1 (bits 95:80)
        (int)(64u << 16),                 // tile_dim0 = 64 (bits 127:112)
        64,                               // tile_dim1 = 64 (bits 143:128)
        1024,                             // tensor_dim0_stride = 1024 elems
        0, 0
    };
    const i32x4 z4 = { 0, 0, 0, 0 };
#if __clang_major__ >= 23
    const i32x8 z8 = { 0, 0, 0, 0, 0, 0, 0, 0 };
    __builtin_amdgcn_tensor_load_to_lds(g0, g1, z4, z4, z8, 0);
#else
    __builtin_amdgcn_tensor_load_to_lds(g0, g1, z4, z4, 0);
#endif
}
#endif

// ---------------------------------------------------------------------------
// Convert W (fp32 [FULL][FULL], row-major) to bf16 (row-major).
// ---------------------------------------------------------------------------
__global__ __launch_bounds__(256)
void convert_w_kernel(const float* __restrict__ W, bf16_t* __restrict__ Wb) {
    const int idx = (blockIdx.x * blockDim.x + threadIdx.x) * 4;
    const float4 f = *(const float4*)(W + idx);
    union { bf16_t h[4]; uint2 u; } p;
    p.h[0] = (bf16_t)f.x; p.h[1] = (bf16_t)f.y;
    p.h[2] = (bf16_t)f.z; p.h[3] = (bf16_t)f.w;
    *(uint2*)(Wb + idx) = p.u;
}

__global__ __launch_bounds__(256)
void zero_r_kernel(uint4* __restrict__ p) {
    p[blockIdx.x * blockDim.x + threadIdx.x] = make_uint4(0u, 0u, 0u, 0u);
}

// ---------------------------------------------------------------------------
// One RNN step:  U[i,j] = lam(i,j)*(sum_k Rin[i,k]*W[j,k] + b[j]) + mask*x[j]
//                Rout   = bf16(relu(U));  last step also writes diag(U) to out.
// A tile: per-lane async-to-LDS.  B tile: one TDM descriptor from wave 0.
// ---------------------------------------------------------------------------
__global__ __launch_bounds__(128)
void rnn_step_kernel(const bf16_t* __restrict__ Wb,
                     const bf16_t* __restrict__ Rin,
                     bf16_t*       __restrict__ Rout,
                     const float*  __restrict__ bias,
                     const float*  __restrict__ xt,
                     float*        __restrict__ out,
                     int is_last) {
    __shared__ bf16_t As[2][BM][BK + PAD];   // A: rows of Rin
    __shared__ bf16_t Bs[2][BN][BK + PAD];   // B: rows of W (B^T view)

    const int tid   = threadIdx.x;
    const int lane  = tid & 31;
    const int wave  = tid >> 5;
    const int i0    = blockIdx.y * BM;
    const int j0    = blockIdx.x * BN;
    const int wr    = (wave >> 1) * 32;
    const int wc    = (wave & 1) * 32;
    const int lhalf = lane >> 4;
    const int lrow  = lane & 15;

    // Cooperative A fill: 64 rows x two 64B half-rows -> 4 b128 per thread.
    const int frow = tid >> 1;
    const int fcol = (tid & 1) * 32;

    v8f acc[2][2] = {};

    auto issue_a = [&](int it, int buf) {   // 4 async ops per wave
        const int ko = it * BK;
        const bf16_t* gA = Rin + (size_t)(i0 + frow) * FULL_SIZE + ko + fcol;
        const uint32_t lA = (uint32_t)(uintptr_t)&As[buf][frow][fcol];
#pragma unroll
        for (int s = 0; s < 4; ++s) async_b128(lA + s * 16, gA + s * 8);
    };
    auto issue_b = [&](int it, int buf) {
        const int ko = it * BK;
#if USE_TDM
        if (wave == 0)                      // 1 TDM op, EXEC-independent DMA
            tdm_load_tile(Wb + (size_t)j0 * FULL_SIZE + ko,
                          (uint32_t)(uintptr_t)&Bs[buf][0][0], FULL_SIZE);
#else
        const bf16_t* gB = Wb + (size_t)(j0 + frow) * FULL_SIZE + ko + fcol;
        const uint32_t lB = (uint32_t)(uintptr_t)&Bs[buf][frow][fcol];
#pragma unroll
        for (int s = 0; s < 4; ++s) async_b128(lB + s * 16, gB + s * 8);
#endif
    };

    issue_a(0, 0);
    issue_b(0, 0);

    for (int it = 0; it < KTILES; ++it) {
        const int cur = it & 1;
        if (it + 1 < KTILES) {
            issue_a(it + 1, cur ^ 1);
            issue_b(it + 1, cur ^ 1);
#if USE_TDM
            __builtin_amdgcn_s_wait_tensorcnt(1);  // cur B landed (wave0; others 0)
            WAIT_ASYNCCNT(4);                      // cur A landed
#else
            WAIT_ASYNCCNT(8);
#endif
        } else {
#if USE_TDM
            __builtin_amdgcn_s_wait_tensorcnt(0);
#endif
            WAIT_ASYNCCNT(0);
        }
        __syncthreads();                           // tile `it` visible to all waves

#pragma unroll
        for (int ks = 0; ks < BK; ks += 32) {
            v16bf afrag[2], bfrag[2];
#pragma unroll
            for (int mf = 0; mf < 2; ++mf) {
                const v8bf lo = *(const v8bf*)&As[cur][wr + mf * 16 + lrow][ks + 16 * lhalf];
                const v8bf hi = *(const v8bf*)&As[cur][wr + mf * 16 + lrow][ks + 16 * lhalf + 8];
                afrag[mf] = __builtin_shufflevector(lo, hi, 0, 1, 2, 3, 4, 5, 6, 7,
                                                    8, 9, 10, 11, 12, 13, 14, 15);
            }
#pragma unroll
            for (int nf = 0; nf < 2; ++nf) {
                const v8bf* p = (const v8bf*)&Bs[cur][wc + nf * 16 + lrow][ks + 16 * lhalf];
                bfrag[nf] = __builtin_shufflevector(p[0], p[1], 0, 1, 2, 3, 4, 5, 6, 7,
                                                    8, 9, 10, 11, 12, 13, 14, 15);
            }
#pragma unroll
            for (int mf = 0; mf < 2; ++mf)
#pragma unroll
                for (int nf = 0; nf < 2; ++nf)
                    acc[mf][nf] = __builtin_amdgcn_wmma_f32_16x16x32_bf16(
                        false, afrag[mf], false, bfrag[nf],
                        (short)0, acc[mf][nf], false, false);
        }
        __syncthreads();   // done reading before buffer refilled at it+2
    }

    // ---- epilogue: bias + Lambda mask + x broadcast + relu; diag on last step
#pragma unroll
    for (int mf = 0; mf < 2; ++mf)
#pragma unroll
        for (int nf = 0; nf < 2; ++nf) {
            const int jn = j0 + wc + nf * 16 + lrow;
            const float bj = bias[jn];
            const float xj = (jn < V_SIZE) ? xt[jn] : 0.0f;
#pragma unroll
            for (int e = 0; e < 8; ++e) {
                const int im = i0 + wr + mf * 16 + e + 8 * lhalf;
                const float wrv = acc[mf][nf][e] + bj;
                float u;
                if (jn >= V_SIZE || jn == im) u = wrv;
                else                          u = 0.5f * (wrv + xj);
                const float r = u > 0.0f ? u : 0.0f;
                Rout[(size_t)im * FULL_SIZE + jn] = (bf16_t)r;
                if (is_last && jn == im) out[im] = u;
            }
        }
}

// ---------------------------------------------------------------------------
// Host launcher. Workspace: Wb bf16 (2MB) | R0 bf16 (1MB) | R1 bf16 (1MB).
// ---------------------------------------------------------------------------
extern "C" void kernel_launch(void* const* d_in, const int* in_sizes, int n_in,
                              void* d_out, int out_size, void* d_ws, size_t ws_size,
                              hipStream_t stream) {
    const float* X = (const float*)d_in[0];
    const float* W = (const float*)d_in[1];
    const float* b = (const float*)d_in[2];
    float* out = (float*)d_out;

    bf16_t* Wb = (bf16_t*)d_ws;
    bf16_t* R0 = (bf16_t*)((char*)d_ws + (size_t)FULL_SIZE * FULL_SIZE * sizeof(bf16_t));
    bf16_t* R1 = R0 + (size_t)V_SIZE * FULL_SIZE;

    convert_w_kernel<<<(FULL_SIZE * FULL_SIZE / 4) / 256, 256, 0, stream>>>(W, Wb);
    zero_r_kernel<<<((size_t)V_SIZE * FULL_SIZE * sizeof(bf16_t) / 16) / 256, 256, 0, stream>>>(
        (uint4*)R0);

    dim3 grid(FULL_SIZE / BN, V_SIZE / BM);   // (16, 8)
    for (int t = 0; t < T_STEPS; ++t) {
        const bf16_t* rin = (t & 1) ? R1 : R0;
        bf16_t* rout      = (t & 1) ? R0 : R1;
        rnn_step_kernel<<<grid, 128, 0, stream>>>(Wb, rin, rout, b,
                                                  X + (size_t)t * V_SIZE, out,
                                                  (t == T_STEPS - 1) ? 1 : 0);
    }
}
